// MPNNetDropDeup_953482740076
// MI455X (gfx1250) — matrix operand order repaired and checked by reading, not compiled, not run on hardware
//
#include <hip/hip_runtime.h>
#include <math.h>

// ---------------------------------------------------------------------------
// CDNA5 (gfx1250) MPNN pipeline.
// Big GEMMs run on v_wmma_f32_16x16x32_bf16 (wave32 WMMA, f32 accumulate),
// strip-mined 16x64 per wave for A-fragment reuse (4 WMMAs / K-step).
// We (edge-conditioned 64x64 matrices, 328MB bf16) is the HBM-bound tensor;
// produced once by a WMMA GEMM, streamed 3x by the LDS-staged message kernel.
// ---------------------------------------------------------------------------

typedef __attribute__((ext_vector_type(16))) __bf16 v16bf;
typedef __attribute__((ext_vector_type(8)))  float  v8f;

#define DEV static __device__ __forceinline__

DEV unsigned short f2bf(float f) {               // RNE float -> bf16 bits
    unsigned u = __float_as_uint(f);
    unsigned r = u + 0x7FFFu + ((u >> 16) & 1u);
    return (unsigned short)(r >> 16);
}
DEV float bf2f(unsigned short u) { return __uint_as_float(((unsigned)u) << 16); }
DEV __bf16 us2bf(unsigned short u) { union { unsigned short s; __bf16 b; } c; c.s = u; return c.b; }
DEV float sigf(float x) { return 1.f / (1.f + expf(-x)); }

// ---------------------------------------------------------------------------
__global__ void k_fill_f32(float* p, int n, float v) {
    int i = blockIdx.x * blockDim.x + threadIdx.x;
    if (i < n) p[i] = v;
}

__global__ void k_f32_to_bf16(const float* __restrict__ s, unsigned short* __restrict__ d, int n) {
    int i = blockIdx.x * blockDim.x + threadIdx.x;
    if (i < n) d[i] = f2bf(s[i]);
}

// out = relu(x @ lin0_w.T + b), keep f32 + bf16 copies
__global__ void k_lin0(const float* __restrict__ x, const float* __restrict__ w,
                       const float* __restrict__ b, float* __restrict__ outf,
                       unsigned short* __restrict__ outb, int N) {
    int i = blockIdx.x * blockDim.x + threadIdx.x;
    if (i >= N * 64) return;
    int n = i >> 6, d = i & 63;
    const float* xr = x + (size_t)n * 14;
    const float* wr = w + (size_t)d * 14;
    float acc = b[d];
#pragma unroll
    for (int k = 0; k < 14; ++k) acc += xr[k] * wr[k];
    acc = fmaxf(acc, 0.f);
    outf[i] = acc;
    outb[i] = f2bf(acc);
}

// hid = relu(edge_attr @ h1_w.T + b) stored bf16 (A operand of the big GEMM)
__global__ void k_edge_mlp1(const float* __restrict__ ea, const float* __restrict__ w,
                            const float* __restrict__ b, unsigned short* __restrict__ hid, int E) {
    int i = blockIdx.x * blockDim.x + threadIdx.x;
    if (i >= E * 128) return;
    int e = i >> 7, j = i & 127;
    const float* a = ea + (size_t)e * 4;
    const float* wr = w + (size_t)j * 4;
    float acc = b[j];
#pragma unroll
    for (int k = 0; k < 4; ++k) acc += a[k] * wr[k];
    hid[i] = f2bf(fmaxf(acc, 0.f));
}

// ---------------------------------------------------------------------------
// bf16 WMMA GEMM: C[M,N] = A[M,K] x B (+bias). 16x64 strip per wave:
// one A fragment feeds 4 v_wmma issues per K-step (A reuse, higher wmma density).
// TRANS_B: B given row-major N x K (weights W[n,k]); else K x N.
// OUT_BF16: store bf16, else f32. K mult of 32; M mult of 16; N mult of 64.
// ---------------------------------------------------------------------------
template <bool TRANS_B, bool OUT_BF16>
__global__ void k_gemm_wmma(const unsigned short* __restrict__ A,
                            const unsigned short* __restrict__ B,
                            void* __restrict__ Cv,
                            const float* __restrict__ bias,
                            int M, int N, int K) {
    int gwave  = (int)((blockIdx.x * blockDim.x + threadIdx.x) >> 5);
    int lane   = threadIdx.x & 31;
    int nstrip = N >> 6;                 // 64-wide output strips
    int mt = gwave / nstrip;
    int ns = gwave - mt * nstrip;
    if (mt * 16 >= M) return;

    int hf = lane >> 4;      // wave half
    int rc = lane & 15;      // row (A) / col (B,D) within tile

    const unsigned short* Arow = A + (size_t)(mt * 16 + rc) * K;

    v8f accs[4];
#pragma unroll
    for (int c = 0; c < 4; ++c) accs[c] = (v8f){0.f,0.f,0.f,0.f,0.f,0.f,0.f,0.f};

    for (int k0 = 0; k0 < K; k0 += 32) {
        v16bf av;
#pragma unroll
        for (int j = 0; j < 16; ++j) {
            // A 16x32 bf16 layout: VGPR0-3 -> K 0..7 (+8 upper half), VGPR4-7 -> K 16..23 (+8)
            av[j] = us2bf(Arow[k0 + j + (j & 8) + (hf << 3)]);
        }
#pragma unroll
        for (int c = 0; c < 4; ++c) {
            int col = (ns << 6) + (c << 4) + rc;
            v16bf bv;
#pragma unroll
            for (int j = 0; j < 16; ++j) {
                // B 32x16 bf16 layout: same K order as A (K 0..7 / 16..23, +8 upper half)
                int kb = k0 + j + (j & 8) + (hf << 3);
                unsigned short bw = TRANS_B ? B[(size_t)col * K + kb]
                                            : B[(size_t)kb * N + col];
                bv[j] = us2bf(bw);
            }
            accs[c] = __builtin_amdgcn_wmma_f32_16x16x32_bf16(false, av, false, bv,
                                                              (short)0, accs[c], false, false);
        }
    }

#pragma unroll
    for (int c = 0; c < 4; ++c) {
        int col = (ns << 6) + (c << 4) + rc;
        float bb = bias ? bias[col] : 0.f;
#pragma unroll
        for (int v = 0; v < 8; ++v) {
            int row = mt * 16 + v + hf * 8;   // D layout: VGPR v -> M = v (+8 upper half)
            float val = accs[c][v] + bb;
            size_t idx = (size_t)row * N + col;
            if (OUT_BF16) ((unsigned short*)Cv)[idx] = f2bf(val);
            else          ((float*)Cv)[idx] = val;
        }
    }
}

// ---------------------------------------------------------------------------
__global__ void k_deg(const int* __restrict__ dst, float* __restrict__ deg, int E) {
    int e = blockIdx.x * blockDim.x + threadIdx.x;
    if (e < E) atomicAdd(&deg[dst[e]], 1.f);
}

// per-edge matvec msg = out[src] @ We[e], scatter-add into agg[dst].
// One 64-thread workgroup per edge. The edge's 8KB We tile is staged into LDS
// with uint4 loads (512B contiguous per wave-issue -> HBM streaming at full
// width), then the FMA loop reads LDS (same-word broadcasts, conflict-free).
__global__ void k_msg(const float* __restrict__ outf, const unsigned short* __restrict__ We,
                      const int* __restrict__ src, const int* __restrict__ dst,
                      float* __restrict__ agg, int E) {
    __shared__ float s[64];
    __shared__ unsigned short sw[4096];        // 8KB: one 64x64 bf16 tile
    int e = blockIdx.x;
    int t = threadIdx.x;
    s[t] = outf[(size_t)src[e] * 64 + t];
    const uint4* gsrc = (const uint4*)(We + (size_t)e * 4096);
    uint4* ldst = (uint4*)sw;
#pragma unroll
    for (int idx = 0; idx < 8; ++idx)          // 512 uint4 total / 64 threads
        ldst[t + idx * 64] = gsrc[t + idx * 64];
    __syncthreads();
    float acc = 0.f;
#pragma unroll 8
    for (int i = 0; i < 64; ++i) acc += s[i] * bf2f(sw[i * 64 + t]);
    atomicAdd(&agg[(size_t)dst[e] * 64 + t], acc);
}

// m = relu(agg * inv_deg + out@root_w + conv_b) -> bf16 (GEMM operand)
__global__ void k_combine(const float* __restrict__ agg, const float* __restrict__ rootm,
                          const float* __restrict__ deg, const float* __restrict__ conv_b,
                          unsigned short* __restrict__ mb, int N) {
    int i = blockIdx.x * blockDim.x + threadIdx.x;
    if (i >= N * 64) return;
    int n = i >> 6, d = i & 63;
    float inv = 1.f / fmaxf(deg[n], 1.f);
    float v = agg[i] * inv + rootm[i] + conv_b[d];
    mb[i] = f2bf(fmaxf(v, 0.f));
}

// GRU gates from gx = m@Wih.T, gh = h@Whh.T (biases added here); h <- update
__global__ void k_gru(const float* __restrict__ gx, const float* __restrict__ gh,
                      const float* __restrict__ bih, const float* __restrict__ bhh,
                      float* __restrict__ outf, unsigned short* __restrict__ outb, int N) {
    int i = blockIdx.x * blockDim.x + threadIdx.x;
    if (i >= N * 64) return;
    int n = i >> 6, d = i & 63;
    const float* gxr = gx + (size_t)n * 192;
    const float* ghr = gh + (size_t)n * 192;
    float xr = gxr[d]       + bih[d];
    float xz = gxr[64 + d]  + bih[64 + d];
    float xn = gxr[128 + d] + bih[128 + d];
    float hr = ghr[d]       + bhh[d];
    float hz = ghr[64 + d]  + bhh[64 + d];
    float hn = ghr[128 + d] + bhh[128 + d];
    float r  = sigf(xr + hr);
    float z  = sigf(xz + hz);
    float nn = tanhf(xn + r * hn);
    float h  = outf[i];
    float hs = (1.f - z) * nn + z * h;
    outf[i] = hs;
    outb[i] = f2bf(hs);
}

// segment boundaries of sorted batch: starts[g] = lower_bound(batch, g)
__global__ void k_starts(const int* __restrict__ batch, int* __restrict__ starts, int N, int G) {
    int g = blockIdx.x * blockDim.x + threadIdx.x;
    if (g > G) return;
    int lo = 0, hi = N;
    while (lo < hi) { int mid = (lo + hi) >> 1; if (batch[mid] < g) lo = mid + 1; else hi = mid; }
    starts[g] = lo;
}

// one set2set step: LSTM cell + per-graph softmax attention + q_star update
// one 64-thread workgroup per graph
__global__ void k_set2set(const float* __restrict__ outf, const int* __restrict__ starts,
                          const float* __restrict__ wih, const float* __restrict__ whh,
                          const float* __restrict__ bih, const float* __restrict__ bhh,
                          float* __restrict__ q_star, float* __restrict__ hl,
                          float* __restrict__ cl, int G) {
    __shared__ float s_q[64];
    __shared__ float s_a[64];
    __shared__ float s_b[64];
    int g = blockIdx.x, t = threadIdx.x;

    // LSTM cell: thread t owns unit t -> gate rows t, 64+t, 128+t, 192+t (i,f,g,o)
    float gi = bih[t]       + bhh[t];
    float gf = bih[64 + t]  + bhh[64 + t];
    float gg = bih[128 + t] + bhh[128 + t];
    float go = bih[192 + t] + bhh[192 + t];
    const float* qs = q_star + (size_t)g * 128;
    for (int k = 0; k < 128; ++k) {
        float q = qs[k];
        gi += q * wih[(size_t)t * 128 + k];
        gf += q * wih[(size_t)(64 + t) * 128 + k];
        gg += q * wih[(size_t)(128 + t) * 128 + k];
        go += q * wih[(size_t)(192 + t) * 128 + k];
    }
    const float* hlr = hl + (size_t)g * 64;
    for (int k = 0; k < 64; ++k) {
        float h = hlr[k];
        gi += h * whh[(size_t)t * 64 + k];
        gf += h * whh[(size_t)(64 + t) * 64 + k];
        gg += h * whh[(size_t)(128 + t) * 64 + k];
        go += h * whh[(size_t)(192 + t) * 64 + k];
    }
    float c_old = cl[(size_t)g * 64 + t];
    float c_new = sigf(gf) * c_old + sigf(gi) * tanhf(gg);
    float h_new = sigf(go) * tanhf(c_new);
    __syncthreads();                       // all hl reads done before writes
    cl[(size_t)g * 64 + t] = c_new;
    hl[(size_t)g * 64 + t] = h_new;
    s_q[t] = h_new;
    __syncthreads();

    // online softmax over this graph's nodes
    int n0 = starts[g], n1 = starts[g + 1];
    float mloc = -3.0e38f, sloc = 0.f;
    for (int n = n0 + t; n < n1; n += 64) {
        const float* o = outf + (size_t)n * 64;
        float e = 0.f;
        for (int k = 0; k < 64; ++k) e += o[k] * s_q[k];
        if (e > mloc) { sloc = sloc * expf(mloc - e) + 1.f; mloc = e; }
        else          { sloc += expf(e - mloc); }
    }
    s_a[t] = mloc; s_b[t] = sloc;
    __syncthreads();
    for (int off = 32; off > 0; off >>= 1) {
        if (t < off) {
            float m1 = s_a[t], m2 = s_a[t + off];
            float M = fmaxf(m1, m2);
            s_b[t] = s_b[t] * expf(m1 - M) + s_b[t + off] * expf(m2 - M);
            s_a[t] = M;
        }
        __syncthreads();
    }
    float emax  = s_a[0];
    float denom = s_b[0] + 1e-16f;
    __syncthreads();

    // weighted read r = sum_n a_n * out[n,:] ; thread t accumulates dim t
    float r_acc = 0.f;
    for (int base = n0; base < n1; base += 64) {
        int n = base + t;
        float w = 0.f;
        if (n < n1) {
            const float* o = outf + (size_t)n * 64;
            float e = 0.f;
            for (int k = 0; k < 64; ++k) e += o[k] * s_q[k];
            w = expf(e - emax) / denom;
        }
        s_a[t] = w;
        __syncthreads();
        int cnt = min(64, n1 - base);
        for (int j = 0; j < cnt; ++j) r_acc += s_a[j] * outf[(size_t)(base + j) * 64 + t];
        __syncthreads();
    }
    q_star[(size_t)g * 128 + t]      = s_q[t];
    q_star[(size_t)g * 128 + 64 + t] = r_acc;
}

// head: emb = relu(q_star @ lin1_w.T + b1); out = emb @ lin2_w[:, :64].T + b2
__global__ void k_head(const float* __restrict__ q_star, const float* __restrict__ w1,
                       const float* __restrict__ b1, const float* __restrict__ w2,
                       const float* __restrict__ b2, float* __restrict__ out, int G) {
    __shared__ float s[64];
    int g = blockIdx.x, t = threadIdx.x;
    const float* qs = q_star + (size_t)g * 128;
    float acc = b1[t];
    for (int k = 0; k < 128; ++k) acc += qs[k] * w1[(size_t)t * 128 + k];
    acc = fmaxf(acc, 0.f);
    s[t] = acc * w2[t];          // lin2_w column 64 hits the zero concat -> skipped
    __syncthreads();
    for (int off = 32; off > 0; off >>= 1) {
        if (t < off) s[t] += s[t + off];
        __syncthreads();
    }
    if (t == 0) out[g] = s[0] + b2[0];
}

// ---------------------------------------------------------------------------
extern "C" void kernel_launch(void* const* d_in, const int* in_sizes, int n_in,
                              void* d_out, int out_size, void* d_ws, size_t ws_size,
                              hipStream_t stream) {
    (void)n_in; (void)out_size; (void)ws_size;

    const float* x         = (const float*)d_in[0];
    const int*   edge_idx  = (const int*)d_in[1];
    const float* edge_attr = (const float*)d_in[2];
    const int*   batch     = (const int*)d_in[3];
    const float* lin0_w = (const float*)d_in[4];  const float* lin0_b = (const float*)d_in[5];
    const float* h1_w   = (const float*)d_in[6];  const float* h1_b   = (const float*)d_in[7];
    const float* h2_w   = (const float*)d_in[8];  const float* h2_b   = (const float*)d_in[9];
    const float* root_w = (const float*)d_in[10]; const float* conv_b = (const float*)d_in[11];
    const float* gru_wih = (const float*)d_in[12]; const float* gru_whh = (const float*)d_in[13];
    const float* gru_bih = (const float*)d_in[14]; const float* gru_bhh = (const float*)d_in[15];
    const float* lstm_wih = (const float*)d_in[16]; const float* lstm_whh = (const float*)d_in[17];
    const float* lstm_bih = (const float*)d_in[18]; const float* lstm_bhh = (const float*)d_in[19];
    const float* lin1_w = (const float*)d_in[20]; const float* lin1_b = (const float*)d_in[21];
    const float* lin2_w = (const float*)d_in[22]; const float* lin2_b = (const float*)d_in[23];

    const int N = in_sizes[0] / 14;
    const int E = in_sizes[2] / 4;
    const int G = 512;
    const int* src = edge_idx;
    const int* dst = edge_idx + E;

    // ----- workspace layout -----
    char* base = (char*)d_ws;
    size_t off = 0;
    auto take = [&](size_t bytes) -> char* {
        char* p = base + off;
        off = (off + bytes + 255) & ~(size_t)255;
        return p;
    };
    float*          out_f  = (float*)take((size_t)N * 64 * 4);
    unsigned short* out_b  = (unsigned short*)take((size_t)N * 64 * 2);
    unsigned short* hid_b  = (unsigned short*)take((size_t)E * 128 * 2);
    unsigned short* We_b   = (unsigned short*)take((size_t)E * 4096 * 2);   // 328 MB bf16
    float*          agg    = (float*)take((size_t)N * 64 * 4);
    float*          deg    = (float*)take((size_t)N * 4);
    float*          rootm  = (float*)take((size_t)N * 64 * 4);
    unsigned short* m_b    = (unsigned short*)take((size_t)N * 64 * 2);
    float*          gx     = (float*)take((size_t)N * 192 * 4);
    float*          gh     = (float*)take((size_t)N * 192 * 4);
    unsigned short* h2w_b  = (unsigned short*)take((size_t)4096 * 128 * 2);
    unsigned short* rotw_b = (unsigned short*)take((size_t)64 * 64 * 2);
    unsigned short* wih_b  = (unsigned short*)take((size_t)192 * 64 * 2);
    unsigned short* whh_b  = (unsigned short*)take((size_t)192 * 64 * 2);
    int*            starts = (int*)take((size_t)(G + 1) * 4);
    float*          q_star = (float*)take((size_t)G * 128 * 4);
    float*          hl     = (float*)take((size_t)G * 64 * 4);
    float*          cl     = (float*)take((size_t)G * 64 * 4);

    auto cvt = [&](const float* s, unsigned short* d, int n) {
        k_f32_to_bf16<<<(n + 255) / 256, 256, 0, stream>>>(s, d, n);
    };
    cvt(h2_w, h2w_b, 4096 * 128);
    cvt(root_w, rotw_b, 64 * 64);
    cvt(gru_wih, wih_b, 192 * 64);
    cvt(gru_whh, whh_b, 192 * 64);

    k_lin0<<<(N * 64 + 255) / 256, 256, 0, stream>>>(x, lin0_w, lin0_b, out_f, out_b, N);
    k_edge_mlp1<<<(E * 128 + 255) / 256, 256, 0, stream>>>(edge_attr, h1_w, h1_b, hid_b, E);

    // We[E,4096] = hid[E,128] @ h2_w[4096,128].T  (bf16 out, f32 accumulate)
    {
        long waves = (long)(E / 16) * (4096 / 64);
        unsigned blocks = (unsigned)((waves + 7) / 8);
        k_gemm_wmma<true, true><<<blocks, 256, 0, stream>>>(hid_b, h2w_b, (void*)We_b, h2_b, E, 4096, 128);
    }

    k_fill_f32<<<(N + 255) / 256, 256, 0, stream>>>(deg, N, 0.f);
    k_deg<<<(E + 255) / 256, 256, 0, stream>>>(dst, deg, E);

    for (int step = 0; step < 3; ++step) {
        k_fill_f32<<<(N * 64 + 255) / 256, 256, 0, stream>>>(agg, N * 64, 0.f);
        k_msg<<<E, 64, 0, stream>>>(out_f, We_b, src, dst, agg, E);
        {   // rootm = out @ root_w
            long waves = (long)(N / 16) * (64 / 64);
            unsigned blocks = (unsigned)((waves + 7) / 8);
            k_gemm_wmma<false, false><<<blocks, 256, 0, stream>>>(out_b, rotw_b, (void*)rootm, nullptr, N, 64, 64);
        }
        k_combine<<<(N * 64 + 255) / 256, 256, 0, stream>>>(agg, rootm, deg, conv_b, m_b, N);
        {   // gx = m @ gru_wih.T ; gh = h @ gru_whh.T
            long waves = (long)(N / 16) * (192 / 64);
            unsigned blocks = (unsigned)((waves + 7) / 8);
            k_gemm_wmma<true, false><<<blocks, 256, 0, stream>>>(m_b,   wih_b, (void*)gx, nullptr, N, 192, 64);
            k_gemm_wmma<true, false><<<blocks, 256, 0, stream>>>(out_b, whh_b, (void*)gh, nullptr, N, 192, 64);
        }
        k_gru<<<(N * 64 + 255) / 256, 256, 0, stream>>>(gx, gh, gru_bih, gru_bhh, out_f, out_b, N);
    }

    k_starts<<<(G + 256) / 256, 256, 0, stream>>>(batch, starts, N, G);
    k_fill_f32<<<(G * 128 + 255) / 256, 256, 0, stream>>>(q_star, G * 128, 0.f);
    k_fill_f32<<<(G * 64 + 255) / 256, 256, 0, stream>>>(hl, G * 64, 0.f);
    k_fill_f32<<<(G * 64 + 255) / 256, 256, 0, stream>>>(cl, G * 64, 0.f);
    for (int step = 0; step < 3; ++step)
        k_set2set<<<G, 64, 0, stream>>>(out_f, starts, lstm_wih, lstm_whh, lstm_bih, lstm_bhh,
                                        q_star, hl, cl, G);

    k_head<<<G, 64, 0, stream>>>(q_star, lin1_w, lin1_b, lin2_w, lin2_b, (float*)d_out, G);
}
